// ClassificationRNN2_53876069761364
// MI455X (gfx1250) — matrix operand order repaired and checked by reading, not compiled, not run on hardware
//
#include <hip/hip_runtime.h>
#include <hip/hip_bf16.h>
#include <math.h>

// Problem dims (match reference)
#define Bb    512
#define Ll    300
#define Dd    128
#define Hh    256
#define FOURH 1024
#define Cc    14
#define Vv    100001

typedef __attribute__((ext_vector_type(16))) __bf16 v16bf;
typedef __attribute__((ext_vector_type(8)))  float  v8f;

union Frag16 {
  v16bf v;
  unsigned int u[8];
  uint4 q[2];
};

__device__ __forceinline__ unsigned short f32_to_bf16(float f) {
  unsigned int u = __float_as_uint(f);
  return (unsigned short)((u + 0x7FFFu + ((u >> 16) & 1u)) >> 16);  // RNE
}

// ---------------------------------------------------------------------------
// One-pass f32 -> bf16 conversion (emb / W_ih / W_hh), once per call.
// ---------------------------------------------------------------------------
__global__ __launch_bounds__(256)
void convert_bf16(const float* __restrict__ src, unsigned short* __restrict__ dst,
                  int n) {
  int i = blockIdx.x * 256 + threadIdx.x;
  if (i < n) dst[i] = f32_to_bf16(src[i]);
}

// ---------------------------------------------------------------------------
// Zero LSTM state: c (f32) and h mirror (bf16)
// ---------------------------------------------------------------------------
__global__ __launch_bounds__(256)
void init_state(float* __restrict__ cstate, unsigned short* __restrict__ hbf) {
  size_t i = (size_t)blockIdx.x * blockDim.x + threadIdx.x;  // 512*256 total
  cstate[i] = 0.0f;
  hbf[i]    = 0;
}

// ---------------------------------------------------------------------------
// Per-timestep fused GEMM:  gacc[512,1024] = [x_t | h] @ [W_ih ; W_hh]^T
// All operands pre-converted bf16; every fragment = two b128 loads.
// One wave computes a 16x64 strip: acc[4], 12 k-steps -> 48 WMMAs.
// Grid (32, 4), block 128 (4 waves); wave w -> n-strip (blockIdx.y*4 + w)*64.
//
// ISA §7.12.2 layouts (wave32):
//   A 16x32 bf16 : lane holds row lane%16; VGPR v holds K pair
//                  (v<4 ? 2v : 16+2(v-4)) + 8*laneHi  -> contiguous groups of 8
//   B 32x16 bf16 : lane holds col lane%16; K order must match A's; code: VGPR v = K pair 2v + 16*laneHi
//                  -> contiguous group of 16, for A as well as B
//   C/D f32      : lane holds col lane%16; VGPR v holds row v + 8*laneHi
// ---------------------------------------------------------------------------
__global__ __launch_bounds__(128)
void lstm_step_gemm(const int*            __restrict__ tokens, // [B, L]
                    const unsigned short* __restrict__ embbf,  // [V, D] bf16
                    const unsigned short* __restrict__ wihbf,  // [4H, D] bf16
                    const unsigned short* __restrict__ whhbf,  // [4H, H] bf16
                    const unsigned short* __restrict__ hbf,    // [B, H] bf16
                    float*                __restrict__ gacc,   // [B, 4H]
                    int t)
{
  const int lane   = threadIdx.x & 31;
  const int wave   = threadIdx.x >> 5;
  const int laneHi = (lane >> 4) & 1;
  const int lr     = lane & 15;

  const int m0 = blockIdx.x * 16;                  // 32 M-tiles (batch)
  const int n0 = (blockIdx.y * 4 + wave) * 64;     // 16 strips of 4 N-tiles

  const int row   = m0 + lr;
  const int token = tokens[row * Ll + t];
  const unsigned short* xrow = embbf + (size_t)token * Dd;
  const unsigned short* hrow = hbf   + (size_t)row   * Hh;

  __builtin_prefetch(xrow, 0, 1);
  __builtin_prefetch(hrow, 0, 1);
  __builtin_prefetch(wihbf + (size_t)(n0 + lr) * Dd, 0, 1);
  __builtin_prefetch(whhbf + (size_t)(n0 + lr) * Hh, 0, 1);

  v8f acc[4] = {};

  // ---- K = 0..127 : x_t (emb gather) x W_ih ----
#pragma unroll
  for (int ks = 0; ks < 4; ++ks) {
    const int k0 = ks * 32;
    Frag16 a;
    a.q[0] = *(const uint4*)(xrow + k0 + 16 * laneHi);
    a.q[1] = *(const uint4*)(xrow + k0 + 16 * laneHi + 8);
#pragma unroll
    for (int i = 0; i < 4; ++i) {
      const unsigned short* wr =
          wihbf + (size_t)(n0 + i * 16 + lr) * Dd + k0 + 16 * laneHi;
      Frag16 b;
      b.q[0] = *(const uint4*)(wr);
      b.q[1] = *(const uint4*)(wr + 8);
      acc[i] = __builtin_amdgcn_wmma_f32_16x16x32_bf16(false, a.v, false, b.v,
                                                       (short)0, acc[i],
                                                       false, false);
    }
  }

  // ---- K = 128..383 : h x W_hh ----
#pragma unroll
  for (int ks = 0; ks < 8; ++ks) {
    const int k0 = ks * 32;
    Frag16 a;
    a.q[0] = *(const uint4*)(hrow + k0 + 16 * laneHi);
    a.q[1] = *(const uint4*)(hrow + k0 + 16 * laneHi + 8);
#pragma unroll
    for (int i = 0; i < 4; ++i) {
      const unsigned short* wr =
          whhbf + (size_t)(n0 + i * 16 + lr) * Hh + k0 + 16 * laneHi;
      Frag16 b;
      b.q[0] = *(const uint4*)(wr);
      b.q[1] = *(const uint4*)(wr + 8);
      acc[i] = __builtin_amdgcn_wmma_f32_16x16x32_bf16(false, a.v, false, b.v,
                                                       (short)0, acc[i],
                                                       false, false);
    }
  }

  // D layout: VGPR v -> row m0 + v + 8*laneHi, col = lr
#pragma unroll
  for (int i = 0; i < 4; ++i) {
    const int ncol = n0 + i * 16 + lr;
#pragma unroll
    for (int v = 0; v < 8; ++v) {
      const int m = m0 + v + (laneHi ? 8 : 0);
      gacc[(size_t)m * FOURH + ncol] = acc[i][v];
    }
  }
}

// ---------------------------------------------------------------------------
// Gate nonlinearities + state update (order i,f,g,o). Writes outputs[:,t,:],
// f32 c-state, and the bf16 h mirror for the next step's WMMA A-operand.
// ---------------------------------------------------------------------------
__global__ __launch_bounds__(256)
void lstm_gate(const float* __restrict__ gacc,   // [B, 4H]
               const float* __restrict__ b_ih,   // [4H]
               const float* __restrict__ b_hh,   // [4H]
               float* __restrict__ cstate,       // [B, H]
               unsigned short* __restrict__ hbf, // [B, H]
               float* __restrict__ outputs,      // [B, L, H]
               int t)
{
  const int b = blockIdx.x;
  const int j = threadIdx.x;
  const float* g = gacc + (size_t)b * FOURH;

  const float gi = g[j]       + b_ih[j]       + b_hh[j];
  const float gf = g[j + 256] + b_ih[j + 256] + b_hh[j + 256];
  const float gg = g[j + 512] + b_ih[j + 512] + b_hh[j + 512];
  const float go = g[j + 768] + b_ih[j + 768] + b_hh[j + 768];

  const float i = 1.0f / (1.0f + expf(-gi));
  const float f = 1.0f / (1.0f + expf(-gf));
  const float o = 1.0f / (1.0f + expf(-go));
  const float gt = tanhf(gg);

  const size_t ci = (size_t)b * Hh + j;
  const float c = f * cstate[ci] + i * gt;
  cstate[ci] = c;
  const float h = o * tanhf(c);

  outputs[((size_t)b * Ll + t) * Hh + j] = h;
  hbf[ci] = f32_to_bf16(h);
}

// ---------------------------------------------------------------------------
// Per-sample attention + MLP + softmax. One block (256 threads) per b.
// Faithful to reference: M[b,h,l] = outputs_flat[b][h*len + l].
// ---------------------------------------------------------------------------
__global__ __launch_bounds__(256)
void attention_head(const int*   __restrict__ lens,
                    const float* __restrict__ outputs, // [B, L, H]
                    const float* __restrict__ W1,      // [H, 2H]
                    const float* __restrict__ b1,      // [H]
                    const float* __restrict__ W2,      // [C, H]
                    const float* __restrict__ b2,      // [C]
                    float* __restrict__ out)           // [B, C]
{
  __shared__ float sh_q[Hh];
  __shared__ float sh_s[Ll + 4];    // scores -> probs
  __shared__ float sh_red[256];
  __shared__ float sh_cat[2 * Hh];  // [ctx | q]
  __shared__ float sh_a[Hh];
  __shared__ float sh_logit[16];

  const int b   = blockIdx.x;
  const int tid = threadIdx.x;
  const int len = lens[b];
  const float* outb = outputs + (size_t)b * Ll * Hh;   // flat [L*H]

  // q = last valid hidden state
  const float qv = outb[(size_t)(len - 1) * Hh + tid];
  sh_q[tid]        = qv;
  sh_cat[Hh + tid] = qv;
  __syncthreads();

  // scores over l (strided across threads)
  float pmax = -INFINITY;
  for (int l = tid; l < Ll; l += 256) {
    float s;
    if (l < len) {
      s = 0.0f;
      for (int h = 0; h < Hh; ++h)
        s += sh_q[h] * outb[(size_t)h * len + l];
    } else {
      s = -INFINITY;
    }
    sh_s[l] = s;
    pmax = fmaxf(pmax, s);
  }
  // block max
  sh_red[tid] = pmax;
  __syncthreads();
  for (int st = 128; st > 0; st >>= 1) {
    if (tid < st) sh_red[tid] = fmaxf(sh_red[tid], sh_red[tid + st]);
    __syncthreads();
  }
  const float mx = sh_red[0];
  __syncthreads();

  // exp + block sum
  float psum = 0.0f;
  for (int l = tid; l < Ll; l += 256) {
    float e = (l < len) ? expf(sh_s[l] - mx) : 0.0f;
    sh_s[l] = e;
    psum += e;
  }
  sh_red[tid] = psum;
  __syncthreads();
  for (int st = 128; st > 0; st >>= 1) {
    if (tid < st) sh_red[tid] += sh_red[tid + st];
    __syncthreads();
  }
  const float tot = sh_red[0];
  __syncthreads();
  for (int l = tid; l < Ll; l += 256) sh_s[l] /= tot;
  __syncthreads();

  // ctx[h] = sum_l prob[l] * outputs[b,l,h]   (thread = h)
  float cv = 0.0f;
  for (int l = 0; l < len; ++l)
    cv += sh_s[l] * outb[(size_t)l * Hh + tid];
  sh_cat[tid] = cv;
  __syncthreads();

  // a = tanh(W1 @ [ctx, q] + b1)   (thread = output j)
  {
    float tacc = b1[tid];
    const float* w1r = W1 + (size_t)tid * (2 * Hh);
    for (int k = 0; k < 2 * Hh; ++k) tacc += w1r[k] * sh_cat[k];
    sh_a[tid] = tanhf(tacc);
  }
  __syncthreads();

  // logits + softmax over C=14
  if (tid < Cc) {
    float tacc = b2[tid];
    const float* w2r = W2 + (size_t)tid * Hh;
    for (int j = 0; j < Hh; ++j) tacc += w2r[j] * sh_a[j];
    sh_logit[tid] = tacc;
  }
  __syncthreads();
  if (tid == 0) {
    float m2 = -INFINITY;
    for (int c = 0; c < Cc; ++c) m2 = fmaxf(m2, sh_logit[c]);
    float s2 = 0.0f;
    for (int c = 0; c < Cc; ++c) s2 += expf(sh_logit[c] - m2);
    sh_red[0] = m2;
    sh_red[1] = s2;
  }
  __syncthreads();
  if (tid < Cc)
    out[(size_t)b * Cc + tid] = expf(sh_logit[tid] - sh_red[0]) / sh_red[1];
}

// ---------------------------------------------------------------------------
extern "C" void kernel_launch(void* const* d_in, const int* in_sizes, int n_in,
                              void* d_out, int out_size, void* d_ws, size_t ws_size,
                              hipStream_t stream) {
  const int*   tokens = (const int*)  d_in[0];   // inputs_arrays [B, L]
  const int*   lens   = (const int*)  d_in[1];   // traj_lens [B]
  const float* emb    = (const float*)d_in[2];   // [V, D]
  const float* W_ih   = (const float*)d_in[3];   // [4H, D]
  const float* W_hh   = (const float*)d_in[4];   // [4H, H]
  const float* b_ih   = (const float*)d_in[5];   // [4H]
  const float* b_hh   = (const float*)d_in[6];   // [4H]
  const float* W1     = (const float*)d_in[7];   // [H, 2H]
  const float* b1     = (const float*)d_in[8];   // [H]
  const float* W2     = (const float*)d_in[9];   // [C, H]
  const float* b2     = (const float*)d_in[10];  // [C]
  float* out = (float*)d_out;

  // Workspace layout (~187 MB), 256B-aligned sections:
  //   outputs [B,L,H] f32 | gacc [B,4H] f32 | cstate [B,H] f32
  //   hbf [B,H] bf16 | embbf [V,D] bf16 | wihbf [4H,D] bf16 | whhbf [4H,H] bf16
  char* ws = (char*)d_ws;
  size_t off = 0;
  auto take = [&](size_t bytes) {
    char* p = ws + off;
    off += (bytes + 255) & ~(size_t)255;
    return p;
  };
  float*          outputs = (float*)take((size_t)Bb * Ll * Hh * 4);
  float*          gacc    = (float*)take((size_t)Bb * FOURH * 4);
  float*          cstate  = (float*)take((size_t)Bb * Hh * 4);
  unsigned short* hbf     = (unsigned short*)take((size_t)Bb * Hh * 2);
  unsigned short* embbf   = (unsigned short*)take((size_t)Vv * Dd * 2);
  unsigned short* wihbf   = (unsigned short*)take((size_t)FOURH * Dd * 2);
  unsigned short* whhbf   = (unsigned short*)take((size_t)FOURH * Hh * 2);

  // One-time (per call) operand conversion to bf16
  {
    const int n_emb = Vv * Dd;            // 12,800,128
    const int n_wih = FOURH * Dd;         // 131,072
    const int n_whh = FOURH * Hh;         // 262,144
    convert_bf16<<<(n_emb + 255) / 256, 256, 0, stream>>>(emb,  embbf, n_emb);
    convert_bf16<<<(n_wih + 255) / 256, 256, 0, stream>>>(W_ih, wihbf, n_wih);
    convert_bf16<<<(n_whh + 255) / 256, 256, 0, stream>>>(W_hh, whhbf, n_whh);
  }

  init_state<<<Bb, 256, 0, stream>>>(cstate, hbf);

  for (int t = 0; t < Ll; ++t) {
    lstm_step_gemm<<<dim3(32, 4), 128, 0, stream>>>(tokens, embbf, wihbf, whhbf,
                                                    hbf, gacc, t);
    lstm_gate<<<Bb, 256, 0, stream>>>(gacc, b_ih, b_hh, cstate, hbf, outputs, t);
  }

  attention_head<<<Bb, 256, 0, stream>>>(lens, outputs, W1, b1, W2, b2, out);
}
